// TwinsBlock_16114717295135
// MI455X (gfx1250) — matrix operand order repaired and checked by reading, not compiled, not run on hardware
//
#include <hip/hip_runtime.h>

// ---------------------------------------------------------------------------
// Twins/Swin LSA block on gfx1250 (MI455X)
//   - all GEMMs via v_wmma_f32_16x16x32_bf16 (wave32)
//   - GEMM A-tiles staged memory->LDS via global_load_async_to_lds_b128
//     (ASYNCcnt path) so 8 waves/block share one A fetch
//   - qkv kept row-major (m x 768) so no div/mod in GEMM epilogue
// ---------------------------------------------------------------------------

typedef __attribute__((ext_vector_type(16))) __bf16 v16bf;
typedef __attribute__((ext_vector_type(8)))  __bf16 bf16x8;
typedef __attribute__((ext_vector_type(8)))  float  v8f;

#define BB    32
#define HH    56
#define WW    56
#define NN    (HH*WW)        // 3136
#define CC    256
#define CC3   768
#define HEADS 8
#define HD    32
#define WS    7
#define TT    49             // tokens per window
#define PP    64             // windows per image
#define HID   1024
#define MROWS (BB*NN)        // 100352 (multiple of 16)
#define SCALE 0.17677669529663687f   // 1/sqrt(32)

// ---- CDNA5 async memory->LDS copy (16B per lane), ASYNCcnt-tracked --------
__device__ __forceinline__ void async_copy_b128(void* lds, const void* g) {
    unsigned l = (unsigned)(unsigned long long)lds;   // low 32 bits = LDS offset
    asm volatile("global_load_async_to_lds_b128 %0, %1, off"
                 :: "v"(l), "v"(g) : "memory");
}
__device__ __forceinline__ void wait_async0() {
    asm volatile("s_wait_asynccnt 0" ::: "memory");
}

// ---- WMMA fragment loaders (ISA 7.12.2 layouts, wave32) -------------------
// A (16x32 bf16, row-major source, row stride ld):
//   lanes 0-15 : M=lane,   V0-3 = K[0..7],  V4-7 = K[16..23]
//   lanes16-31 : M=lane-16,V0-3 = K[8..15], V4-7 = K[24..31]
__device__ __forceinline__ v16bf load_fragA(const __bf16* base, int ld, int row0, int lane) {
    int m    = row0 + (lane & 15);
    int half = lane >> 4;
    const __bf16* p = base + m * ld + half * 8;
    union { v16bf v; bf16x8 h[2]; } u;
    u.h[0] = *(const bf16x8*)(p);
    u.h[1] = *(const bf16x8*)(p + 16);
    return u.v;
}
// B (32x16 bf16) loaded from an N-major ("Bt[n][k]") source, row stride ld:
//   lanes 0-15 : N=lane,    V0-3 = K[0..7],  V4-7 = K[16..23]
//   lanes16-31 : N=lane-16, V0-3 = K[8..15], V4-7 = K[24..31]
__device__ __forceinline__ v16bf load_fragB(const __bf16* base, int ld, int col0, int lane) {
    int n    = col0 + (lane & 15);
    int half = lane >> 4;
    const __bf16* p = base + n * ld + half * 8;
    union { v16bf v; bf16x8 h[2]; } u;
    u.h[0] = *(const bf16x8*)(p);
    u.h[1] = *(const bf16x8*)(p + 16);
    return u.v;
}
__device__ __forceinline__ v8f wmma_bf16(v16bf a, v16bf b, v8f c) {
    return __builtin_amdgcn_wmma_f32_16x16x32_bf16(false, a, false, b, (short)0, c, false, false);
}

// ---- weight convert + transpose: wt[n*K+k] = (bf16)w[k*N+n] ---------------
__global__ void k_wt_transpose(const float* __restrict__ w, __bf16* __restrict__ wt,
                               int K, int N) {
    long i = (long)blockIdx.x * blockDim.x + threadIdx.x;
    if (i < (long)K * N) {
        long n = i / K, k = i % K;
        wt[i] = (__bf16)w[k * (long)N + n];
    }
}

// ---- LN1 + window partition: x (B,N,C) f32 -> xw (B,P,49,C) bf16 ----------
__global__ __launch_bounds__(256) void k_ln1_window(const float* __restrict__ x,
                                                    const float* __restrict__ g,
                                                    const float* __restrict__ bta,
                                                    __bf16* __restrict__ xw) {
    int wid = threadIdx.x >> 5, lane = threadIdx.x & 31;
    long row = (long)blockIdx.x * 8 + wid;           // < MROWS
    const float* xp = x + row * CC + lane * 8;
    float v[8]; float s = 0.f, s2 = 0.f;
#pragma unroll
    for (int i = 0; i < 8; i++) { v[i] = xp[i]; s += v[i]; s2 += v[i]*v[i]; }
#pragma unroll
    for (int o = 16; o > 0; o >>= 1) { s += __shfl_xor(s, o, 32); s2 += __shfl_xor(s2, o, 32); }
    float mean = s * (1.f/CC);
    float inv  = rsqrtf(s2 * (1.f/CC) - mean*mean + 1e-5f);
    int bi = (int)(row / NN), n = (int)(row % NN);
    int rh = n / WW, rw = n % WW;
    int p = (rh/WS)*8 + (rw/WS);
    int t = (rh%WS)*WS + (rw%WS);
    __bf16* op = xw + (((long)bi*PP + p)*TT + t)*CC + lane*8;
#pragma unroll
    for (int i = 0; i < 8; i++) {
        int c = lane*8 + i;
        op[i] = (__bf16)((v[i]-mean)*inv*g[c] + bta[c]);
    }
}

// ---- LN2 (no permute): out1 f32 -> h2 bf16 --------------------------------
__global__ __launch_bounds__(256) void k_ln2(const float* __restrict__ x,
                                             const float* __restrict__ g,
                                             const float* __restrict__ bta,
                                             __bf16* __restrict__ h2) {
    int wid = threadIdx.x >> 5, lane = threadIdx.x & 31;
    long row = (long)blockIdx.x * 8 + wid;
    const float* xp = x + row * CC + lane * 8;
    float v[8]; float s = 0.f, s2 = 0.f;
#pragma unroll
    for (int i = 0; i < 8; i++) { v[i] = xp[i]; s += v[i]; s2 += v[i]*v[i]; }
#pragma unroll
    for (int o = 16; o > 0; o >>= 1) { s += __shfl_xor(s, o, 32); s2 += __shfl_xor(s2, o, 32); }
    float mean = s * (1.f/CC);
    float inv  = rsqrtf(s2 * (1.f/CC) - mean*mean + 1e-5f);
    __bf16* op = h2 + row * CC + lane * 8;
#pragma unroll
    for (int i = 0; i < 8; i++) {
        int c = lane*8 + i;
        op[i] = (__bf16)((v[i]-mean)*inv*g[c] + bta[c]);
    }
}

// ---- QKV GEMM: block owns a 16-row M-tile (A staged in LDS once),
//      8 waves sweep 48 N-tiles (6 each); row-major bf16 output -------------
__global__ __launch_bounds__(256) void k_gemm_qkv(const __bf16* __restrict__ A,
                                                  const __bf16* __restrict__ Bt,
                                                  const float* __restrict__ bias,
                                                  __bf16* __restrict__ qkv) {
    __shared__ __align__(16) __bf16 atile[16*CC];   // 8 KB
    int tid = threadIdx.x, lane = tid & 31, wid = tid >> 5;
    int m0 = blockIdx.x * 16;
    {   // async-stage contiguous 8KB A tile
        const char* gs = (const char*)(A + (long)m0*CC);
        char* ld = (char*)atile;
#pragma unroll
        for (int i = 0; i < 2; i++) {
            int off = (tid + i*256) * 16;
            async_copy_b128(ld + off, gs + off);
        }
    }
    wait_async0();
    __syncthreads();
    v16bf afr[8];
#pragma unroll
    for (int s = 0; s < 8; s++) afr[s] = load_fragA(atile + s*32, CC, 0, lane);
#pragma unroll
    for (int ntl = 0; ntl < 6; ntl++) {
        int n0 = (wid*6 + ntl)*16;
        v8f acc = {0.f,0.f,0.f,0.f,0.f,0.f,0.f,0.f};
#pragma unroll
        for (int s = 0; s < 8; s++) {
            __builtin_prefetch(Bt + (long)n0*CC + s*32 + 256, 0, 1);
            v16bf b = load_fragB(Bt + s*32, CC, n0, lane);
            acc = wmma_bf16(afr[s], b, acc);
        }
        int n = n0 + (lane & 15);
        float bv = bias[n];
        long mb = m0 + (lane >> 4) * 8;
        __bf16* op = qkv + mb*CC3 + n;
#pragma unroll
        for (int r = 0; r < 8; r++) op[r*CC3] = (__bf16)(acc[r] + bv);
    }
}

// ---- Attention per (b, window, head): softmax(q k^T) v, fused unwindow+res
// qkv is row-major: row = (b*64+p)*49 + t, cols [0,256)=q [256,512)=k [512,768)=v
__global__ __launch_bounds__(32) void k_attn(const __bf16* __restrict__ qkv,
                                             const float* __restrict__ x,
                                             float* __restrict__ out1) {
    __shared__ __align__(16) __bf16 qs[64*32];
    __shared__ __align__(16) __bf16 ks[64*32];
    __shared__ __align__(16) __bf16 vt[32*64];   // transposed v: [d][t]
    __shared__ __align__(16) __bf16 ps[64*64];   // softmax probs
    __shared__ __align__(16) float  ss[64*64];   // scores
    int lane = threadIdx.x;
    int idx = blockIdx.x;                  // b*512 + p*8 + h
    int h = idx & 7, p = (idx >> 3) & 63, b = idx >> 9;
    int w = b*PP + p;
    // stage q (scaled), k, v^T with zero padding to 64 tokens
    for (int rr = 0; rr < 2; rr++) {
        int t = lane + rr*32;
        if (t < TT) {
            const __bf16* rowb = qkv + ((long)w*TT + t)*CC3 + h*HD;
            const bf16x8* qr = (const bf16x8*)(rowb);
            const bf16x8* kr = (const bf16x8*)(rowb + CC);
            const bf16x8* vr = (const bf16x8*)(rowb + 2*CC);
            bf16x8* qd = (bf16x8*)(qs + t*32);
            bf16x8* kd = (bf16x8*)(ks + t*32);
#pragma unroll
            for (int c = 0; c < 4; c++) {
                bf16x8 qv = qr[c];
#pragma unroll
                for (int e = 0; e < 8; e++) qv[e] = (__bf16)((float)qv[e] * SCALE);
                qd[c] = qv;
                kd[c] = kr[c];
                bf16x8 vv = vr[c];
#pragma unroll
                for (int e = 0; e < 8; e++) vt[(c*8+e)*64 + t] = vv[e];
            }
        } else {
#pragma unroll
            for (int d = 0; d < HD; d++) {
                qs[t*32+d] = (__bf16)0.f; ks[t*32+d] = (__bf16)0.f; vt[d*64+t] = (__bf16)0.f;
            }
        }
    }
    __syncthreads();
    // S = q @ k^T  (4x4 tiles, K=32 single WMMA each)
#pragma unroll
    for (int mt = 0; mt < 4; mt++)
#pragma unroll
        for (int nt = 0; nt < 4; nt++) {
            v16bf a = load_fragA(qs, 32, mt*16, lane);
            v16bf bb = load_fragB(ks, 32, nt*16, lane);
            v8f acc = {0.f,0.f,0.f,0.f,0.f,0.f,0.f,0.f};
            acc = wmma_bf16(a, bb, acc);
            int n = nt*16 + (lane & 15);
            int mb = mt*16 + (lane >> 4)*8;
#pragma unroll
            for (int r = 0; r < 8; r++) ss[(mb+r)*64 + n] = acc[r];
        }
    __syncthreads();
    // row softmax over 49 valid columns -> ps (bf16), mask/zero the rest
    for (int rr = 0; rr < 2; rr++) {
        int t = lane + rr*32;
        if (t < TT) {
            float mx = -1e30f;
            for (int j = 0; j < TT; j++) mx = fmaxf(mx, ss[t*64+j]);
            float sum = 0.f;
            for (int j = 0; j < TT; j++) { float e = __expf(ss[t*64+j]-mx); ss[t*64+j] = e; sum += e; }
            float inv = 1.f / sum;
            for (int j = 0; j < TT; j++) ps[t*64+j] = (__bf16)(ss[t*64+j]*inv);
            for (int j = TT; j < 64; j++) ps[t*64+j] = (__bf16)0.f;
        } else {
            for (int j = 0; j < 64; j++) ps[t*64+j] = (__bf16)0.f;
        }
    }
    __syncthreads();
    // out = P @ v (4x2 tiles, K=64 in 2 steps); fused un-window + residual add
#pragma unroll
    for (int mt = 0; mt < 4; mt++)
#pragma unroll
        for (int nt = 0; nt < 2; nt++) {
            v8f acc = {0.f,0.f,0.f,0.f,0.f,0.f,0.f,0.f};
#pragma unroll
            for (int k = 0; k < 64; k += 32) {
                v16bf a = load_fragA(ps + k, 64, mt*16, lane);
                v16bf bb = load_fragB(vt + k, 64, nt*16, lane);
                acc = wmma_bf16(a, bb, acc);
            }
            int n = nt*16 + (lane & 15);          // d within head
            int mb = mt*16 + (lane >> 4)*8;
#pragma unroll
            for (int r = 0; r < 8; r++) {
                int t = mb + r;
                if (t < TT) {
                    int gh = (p >> 3)*WS + t/WS;
                    int gw = (p & 7)*WS + t%WS;
                    long off = (((long)b*NN) + gh*WW + gw)*CC + h*HD + n;
                    out1[off] = x[off] + acc[r];
                }
            }
        }
}

// ---- FC1 + exact GELU + MLP LayerNorm, fused per 16-row tile --------------
__global__ __launch_bounds__(256) void k_fc1_gelu_ln(const __bf16* __restrict__ A,
                                                     const __bf16* __restrict__ Bt,
                                                     const float* __restrict__ bias,
                                                     const float* __restrict__ mg,
                                                     const float* __restrict__ mbv,
                                                     __bf16* __restrict__ hln) {
    __shared__ __align__(16) __bf16 atile[16*CC];   // 8 KB
    __shared__ __align__(16) __bf16 tile[16*HID];   // 32 KB
    __shared__ float redS[16][17], redS2[16][17];
    int tid = threadIdx.x, lane = tid & 31, wid = tid >> 5;
    int m0 = blockIdx.x * 16;
    {   // async-stage contiguous 8KB A tile
        const char* gs = (const char*)(A + (long)m0*CC);
        char* ld = (char*)atile;
#pragma unroll
        for (int i = 0; i < 2; i++) {
            int off = (tid + i*256) * 16;
            async_copy_b128(ld + off, gs + off);
        }
    }
    wait_async0();
    __syncthreads();
    v16bf afr[8];
#pragma unroll
    for (int s = 0; s < 8; s++) afr[s] = load_fragA(atile + s*32, CC, 0, lane);
#pragma unroll
    for (int ntl = 0; ntl < 8; ntl++) {
        int n0 = (wid*8 + ntl)*16;
        v8f acc = {0.f,0.f,0.f,0.f,0.f,0.f,0.f,0.f};
#pragma unroll
        for (int s = 0; s < 8; s++) {
            v16bf bfr = load_fragB(Bt + s*32, CC, n0, lane);
            acc = wmma_bf16(afr[s], bfr, acc);
        }
        int n = n0 + (lane & 15);
        float bv = bias[n];
        int mb = (lane >> 4)*8;
#pragma unroll
        for (int r = 0; r < 8; r++) {
            float v = acc[r] + bv;
            float gl = 0.5f * v * (1.0f + erff(v * 0.70710678118654752f));
            tile[(mb+r)*HID + n] = (__bf16)gl;
        }
    }
    __syncthreads();
    int row = tid >> 4;     // 0..15
    int j   = tid & 15;     // 0..15
    float s = 0.f, s2 = 0.f;
    for (int c = j; c < HID; c += 16) { float v = (float)tile[row*HID+c]; s += v; s2 += v*v; }
    redS[row][j] = s; redS2[row][j] = s2;
    __syncthreads();
    float ts = 0.f, ts2 = 0.f;
#pragma unroll
    for (int q = 0; q < 16; q++) { ts += redS[row][q]; ts2 += redS2[row][q]; }
    float mean = ts * (1.f/HID);
    float inv  = rsqrtf(ts2 * (1.f/HID) - mean*mean + 1e-5f);
    __bf16* op = hln + ((long)m0 + row)*HID;
    for (int c = j; c < HID; c += 16) {
        float v = (float)tile[row*HID+c];
        op[c] = (__bf16)((v-mean)*inv*mg[c] + mbv[c]);
    }
}

// ---- FC2 GEMM: A tile (16x1024, 32KB) async-staged to LDS; 8 waves cover
//      16 N-tiles (2 each); epilogue adds bias + residual -> f32 output -----
__global__ __launch_bounds__(256) void k_gemm_fc2(const __bf16* __restrict__ A,
                                                  const __bf16* __restrict__ Bt,
                                                  const float* __restrict__ bias,
                                                  const float* __restrict__ res,
                                                  float* __restrict__ out) {
    __shared__ __align__(16) __bf16 atile[16*HID];  // 32 KB
    int tid = threadIdx.x, lane = tid & 31, wid = tid >> 5;
    int m0 = blockIdx.x * 16;
    {   // async-stage contiguous 32KB A tile
        const char* gs = (const char*)(A + (long)m0*HID);
        char* ld = (char*)atile;
#pragma unroll
        for (int i = 0; i < 8; i++) {
            int off = (tid + i*256) * 16;
            async_copy_b128(ld + off, gs + off);
        }
    }
    wait_async0();
    __syncthreads();
#pragma unroll
    for (int ntl = 0; ntl < 2; ntl++) {
        int n0 = (wid*2 + ntl)*16;
        v8f acc = {0.f,0.f,0.f,0.f,0.f,0.f,0.f,0.f};
#pragma unroll 8
        for (int k = 0; k < HID; k += 32) {
            v16bf a = load_fragA(atile + k, HID, 0, lane);
            v16bf b = load_fragB(Bt + k, HID, n0, lane);
            acc = wmma_bf16(a, b, acc);
        }
        int n = n0 + (lane & 15);
        float bv = bias[n];
        int mb = m0 + (lane >> 4)*8;
#pragma unroll
        for (int r = 0; r < 8; r++) {
            long off = (long)(mb+r)*CC + n;
            out[off] = acc[r] + bv + res[off];
        }
    }
}

// ---------------------------------------------------------------------------
extern "C" void kernel_launch(void* const* d_in, const int* in_sizes, int n_in,
                              void* d_out, int out_size, void* d_ws, size_t ws_size,
                              hipStream_t stream) {
    const float* x     = (const float*)d_in[0];
    const float* ln1_g = (const float*)d_in[1];
    const float* ln1_b = (const float*)d_in[2];
    const float* qkv_w = (const float*)d_in[3];
    const float* qkv_b = (const float*)d_in[4];
    const float* ln2_g = (const float*)d_in[5];
    const float* ln2_b = (const float*)d_in[6];
    const float* fc1_w = (const float*)d_in[7];
    const float* fc1_b = (const float*)d_in[8];
    const float* mln_g = (const float*)d_in[9];
    const float* mln_b = (const float*)d_in[10];
    const float* fc2_w = (const float*)d_in[11];
    const float* fc2_b = (const float*)d_in[12];
    float* out = (float*)d_out;

    // workspace layout (regions reused across phases)
    const size_t OFF_A = 0;                                   // xw bf16 -> hln bf16
    const size_t OFF_B = OFF_A + (size_t)MROWS*HID*2;         // qkv bf16 -> h2 bf16
    const size_t OFF_C = OFF_B + (size_t)MROWS*CC3*2;         // out1 f32
    const size_t OFF_W = OFF_C + (size_t)MROWS*CC*4;          // bf16 weights
    char* ws = (char*)d_ws;
    __bf16* xw     = (__bf16*)(ws + OFF_A);
    __bf16* hln    = (__bf16*)(ws + OFF_A);
    __bf16* qkv    = (__bf16*)(ws + OFF_B);
    __bf16* h2     = (__bf16*)(ws + OFF_B);
    float*  out1   = (float*)(ws + OFF_C);
    __bf16* qkv_wt = (__bf16*)(ws + OFF_W);
    __bf16* fc1_wt = qkv_wt + 768*256;
    __bf16* fc2_wt = fc1_wt + 1024*256;

    k_wt_transpose<<<(768*256 + 255)/256, 256, 0, stream>>>(qkv_w, qkv_wt, 256, 768);
    k_wt_transpose<<<(1024*256 + 255)/256, 256, 0, stream>>>(fc1_w, fc1_wt, 256, 1024);
    k_wt_transpose<<<(1024*256 + 255)/256, 256, 0, stream>>>(fc2_w, fc2_wt, 1024, 256);

    k_ln1_window<<<MROWS/8, 256, 0, stream>>>(x, ln1_g, ln1_b, xw);
    k_gemm_qkv<<<MROWS/16, 256, 0, stream>>>(xw, qkv_wt, qkv_b, qkv);
    k_attn<<<BB*PP*HEADS, 32, 0, stream>>>(qkv, x, out1);
    k_ln2<<<MROWS/8, 256, 0, stream>>>(out1, ln2_g, ln2_b, h2);
    k_fc1_gelu_ln<<<MROWS/16, 256, 0, stream>>>(h2, fc1_wt, fc1_b, mln_g, mln_b, hln);
    k_gemm_fc2<<<MROWS/16, 256, 0, stream>>>(hln, fc2_wt, fc2_b, out1, out);
}
